// MultiHeadAttention_3779571220507
// MI455X (gfx1250) — compile-verified, matrix operand order checked by reading, not run on hardware
//
#include <hip/hip_runtime.h>
#include <hip/hip_bf16.h>

#define DMODEL 1024
#define NHEADS 16
#define DK     64
#define BATCH  2
#define SEQ    2048

typedef __attribute__((ext_vector_type(16))) __bf16       v16bf;
typedef __attribute__((ext_vector_type(8)))  float        v8f;
typedef __attribute__((ext_vector_type(4)))  unsigned int u32x4;
typedef __attribute__((ext_vector_type(2)))  unsigned int u32x2;
typedef __attribute__((ext_vector_type(4)))  float        f32x4;
typedef __attribute__((ext_vector_type(8)))  int          i32x8;
typedef __attribute__((ext_vector_type(4)))  int          i32x4;

union BF16Frag { v16bf v; u32x4 q[2]; unsigned short h[16]; };
union F32Acc   { v8f   v; float f[8]; };

#if __has_builtin(__builtin_amdgcn_tensor_load_to_lds) && __has_builtin(__builtin_amdgcn_s_wait_tensorcnt)
#define USE_TDM 1
#else
#define USE_TDM 0
#endif

__device__ __forceinline__ unsigned short f2bf(float x) {
  unsigned u = __float_as_uint(x);
  u += 0x7FFFu + ((u >> 16) & 1u);      // round-to-nearest-even
  return (unsigned short)(u >> 16);
}
__device__ __forceinline__ unsigned pack2bf(float a, float b) {
  return (unsigned)f2bf(a) | ((unsigned)f2bf(b) << 16);
}

#if USE_TDM
// Issue one 2D TDM tile load (bf16 elements): tile_x contiguous elements per row,
// tile_y rows, row stride stride_x elements; LDS padded by pad_amt(code) DWORDs
// every pad_int(code) DWORDs. D# bit layout per CDNA5 ISA ch.8.
// This toolchain declares the 6-arg form: (g0 x4, g1 x8, g2 x4, g3 x4, x8, cpol).
__device__ __forceinline__ void tdm_load_2d(unsigned lds_byte_addr, const void* gaddr,
                                            unsigned tile_x, unsigned tile_y,
                                            unsigned stride_x,
                                            unsigned pad_int_code, unsigned pad_amt_code) {
  const unsigned long long ga = (unsigned long long)gaddr;
  const unsigned tensor_x = 1u << 30, tensor_y = 1u << 20;  // never clip
  u32x4 g0;
  g0.x = 1u;                                               // count=1, user mode
  g0.y = lds_byte_addr;
  g0.z = (unsigned)ga;
  g0.w = (unsigned)((ga >> 32) & 0x01FFFFFFu) | (2u << 30); // addr[56:32] | type=2
  i32x8 g1;
  g1[0] = (int)((1u << 16) | (1u << 20) | (pad_int_code << 22) | (pad_amt_code << 25));
  g1[1] = (int)((tensor_x & 0xFFFFu) << 16);               // abar=0 | tensor_dim0.lo
  g1[2] = (int)((tensor_x >> 16) | ((tensor_y & 0xFFFFu) << 16));
  g1[3] = (int)((tensor_y >> 16) | (tile_x << 16));        // tensor_dim1.hi | tile_dim0
  g1[4] = (int)(tile_y & 0xFFFFu);                         // tile_dim1, tile_dim2=0
  g1[5] = (int)stride_x;                                   // tensor_dim0_stride[31:0]
  g1[6] = 0;
  g1[7] = 0;
  const i32x4 z4 = {0, 0, 0, 0};
  const i32x8 z8 = {0, 0, 0, 0, 0, 0, 0, 0};
  __builtin_amdgcn_tensor_load_to_lds(g0, g1, z4, z4, z8, 0);
}
__device__ __forceinline__ unsigned lds_addr_of(const void* p) {
  return (unsigned)(unsigned long long)p;  // LDS aperture: low 32 bits = LDS offset
}
#endif

// 16x32 bf16 A/B fragment from LDS (row-major, stride in halves).
// lane l (g=l>>4, r=l&15): halves 0..7 -> K=8g+h, halves 8..15 -> K=16+8g+(h-8).
__device__ __forceinline__ BF16Frag load_frag(const unsigned short* s, int stride, int lane) {
  BF16Frag f;
  const int g = lane >> 4, r = lane & 15;
  const unsigned short* p = s + r * stride + 8 * g;
  f.q[0] = *(const u32x4*)(p);
  f.q[1] = *(const u32x4*)(p + 16);
  return f;
}

__device__ __forceinline__ v8f wmma_bf16(const BF16Frag& a, const BF16Frag& b, v8f c) {
  return __builtin_amdgcn_wmma_f32_16x16x32_bf16(false, a.v, false, b.v, (short)0, c, false, false);
}

// ---------------------------------------------------------------- weights f32 -> bf16
__global__ __launch_bounds__(256)
void cvt_bf16_kernel(const float* __restrict__ src, unsigned short* __restrict__ dst, int n) {
  const int i = (blockIdx.x * 256 + threadIdx.x) * 4;
  if (i + 3 < n) {
    f32x4 v = *(const f32x4*)(src + i);
    u32x2 o;
    o.x = pack2bf(v.x, v.y);
    o.y = pack2bf(v.z, v.w);
    *(u32x2*)(dst + i) = o;
  }
}

// ---------------------------------------------------------------- GEMM: C = A @ W^T + bias
// A: [M=4096, K=1024] (f32 or bf16). W bf16 [N][K] row-major.
// OUT_MODE 0: f32 [M,N]. 1: bf16 [B,H,S,DK]. 2: bf16 [B,H,DK,S] (transposed, b128 stores).
template <bool A_F32, int OUT_MODE>
__global__ __launch_bounds__(256)
void gemm_kernel(const void* __restrict__ Aptr,
                 const unsigned short* __restrict__ Wbf,
                 const float* __restrict__ bias,
                 void* __restrict__ Cptr) {
  constexpr int K = DMODEL, N = DMODEL;
  constexpr int LDA = 40;  // 32 + 8 halves pad (TDM: 16 DWORDs + 4 DWORDs pad)
  __shared__ __align__(16) unsigned short Asm[128 * LDA];
  __shared__ __align__(16) unsigned short Bsm[128 * LDA];

  const int tid  = threadIdx.x;
  const int wave = tid >> 5, lane = tid & 31;
  const int bm = blockIdx.y * 128;
  const int bn = blockIdx.x * 128;
  const int wm = (wave & 3) * 32;
  const int wn = (wave >> 2) * 64;

  F32Acc acc[2][4];
#pragma unroll
  for (int mi = 0; mi < 2; ++mi)
#pragma unroll
    for (int ni = 0; ni < 4; ++ni)
#pragma unroll
      for (int i = 0; i < 8; ++i) acc[mi][ni].f[i] = 0.f;

  const int lr = tid >> 1;  // 0..127 tile row
  const int lh = tid & 1;

  for (int k0 = 0; k0 < K; k0 += 32) {
    // ---- A tile
    if (A_F32) {
      const float* A = (const float*)Aptr;
      const float* ap = A + (size_t)(bm + lr) * K + k0 + 16 * lh;
      unsigned short* as = Asm + lr * LDA + 16 * lh;
      __builtin_prefetch(ap + 32, 0, 0);   // next K slab -> global_prefetch_b8
#pragma unroll
      for (int j = 0; j < 4; ++j) {
        f32x4 v = *(const f32x4*)(ap + 4 * j);
        u32x2 o;
        o.x = pack2bf(v.x, v.y);
        o.y = pack2bf(v.z, v.w);
        *(u32x2*)(as + 4 * j) = o;
      }
    } else {
#if USE_TDM
      if (wave == 6) {
        tdm_load_2d(lds_addr_of(Asm), (const unsigned short*)Aptr + (size_t)bm * K + k0,
                    /*tile_x=*/32, /*tile_y=*/128, /*stride=*/K, /*pi=16dw*/3, /*pa=4dw*/3);
        __builtin_amdgcn_s_wait_tensorcnt(0);
      }
#else
      const unsigned short* A = (const unsigned short*)Aptr;
      const unsigned short* ap = A + (size_t)(bm + lr) * K + k0 + 16 * lh;
      unsigned short* as = Asm + lr * LDA + 16 * lh;
      *(u32x4*)(as)     = *(const u32x4*)(ap);
      *(u32x4*)(as + 8) = *(const u32x4*)(ap + 8);
#endif
    }
    // ---- B (weight) tile
#if USE_TDM
    if (wave == 7) {
      tdm_load_2d(lds_addr_of(Bsm), Wbf + (size_t)bn * K + k0,
                  /*tile_x=*/32, /*tile_y=*/128, /*stride=*/K, 3, 3);
      __builtin_amdgcn_s_wait_tensorcnt(0);
    }
#else
    {
      const unsigned short* wp = Wbf + (size_t)(bn + lr) * K + k0 + 16 * lh;
      unsigned short* bs = Bsm + lr * LDA + 16 * lh;
      *(u32x4*)(bs)     = *(const u32x4*)(wp);
      *(u32x4*)(bs + 8) = *(const u32x4*)(wp + 8);
    }
#endif
    __syncthreads();

    BF16Frag af[2], bfw[4];
#pragma unroll
    for (int mi = 0; mi < 2; ++mi) af[mi] = load_frag(Asm + (wm + 16 * mi) * LDA, LDA, lane);
#pragma unroll
    for (int ni = 0; ni < 4; ++ni) bfw[ni] = load_frag(Bsm + (wn + 16 * ni) * LDA, LDA, lane);
#pragma unroll
    for (int mi = 0; mi < 2; ++mi)
#pragma unroll
      for (int ni = 0; ni < 4; ++ni)
        acc[mi][ni].v = wmma_bf16(af[mi], bfw[ni], acc[mi][ni].v);
    __syncthreads();
  }

  const int g = lane >> 4, c = lane & 15;
#pragma unroll
  for (int mi = 0; mi < 2; ++mi) {
#pragma unroll
    for (int ni = 0; ni < 4; ++ni) {
      const int n = bn + wn + 16 * ni + c;
      const float bv = bias[n];
      const int m0 = bm + wm + 16 * mi + 8 * g;  // 8 consecutive rows m0..m0+7
      if (OUT_MODE == 2) {
        // v^T layout [B,H,DK,S]: 8 consecutive s -> one b128 store
        const int b = m0 >> 11, s0 = m0 & 2047;
        const int h = n >> 6, d = n & 63;
        u32x4 o;
        o.x = pack2bf(acc[mi][ni].f[0] + bv, acc[mi][ni].f[1] + bv);
        o.y = pack2bf(acc[mi][ni].f[2] + bv, acc[mi][ni].f[3] + bv);
        o.z = pack2bf(acc[mi][ni].f[4] + bv, acc[mi][ni].f[5] + bv);
        o.w = pack2bf(acc[mi][ni].f[6] + bv, acc[mi][ni].f[7] + bv);
        unsigned short* C = (unsigned short*)Cptr;
        *(u32x4*)(C + ((size_t)((b * NHEADS + h) * DK + d)) * SEQ + s0) = o;
      } else {
#pragma unroll
        for (int i = 0; i < 8; ++i) {
          const int m = m0 + i;
          const float val = acc[mi][ni].f[i] + bv;
          if (OUT_MODE == 1) {
            unsigned short* C = (unsigned short*)Cptr;
            const int b = m >> 11, s = m & 2047;
            const int h = n >> 6, d = n & 63;
            C[((size_t)((b * NHEADS + h) * SEQ + s)) * DK + d] = f2bf(val);
          } else {
            float* C = (float*)Cptr;
            C[(size_t)m * N + n] = val;
          }
        }
      }
    }
  }
}

// ---------------------------------------------------------------- flash attention
// q/k bf16 [B,H,S,DK]; vT bf16 [B,H,DK,S]; ctx bf16 [B,S,DMODEL].
__global__ __launch_bounds__(256)
void attn_kernel(const unsigned short* __restrict__ qb,
                 const unsigned short* __restrict__ kbuf,
                 const unsigned short* __restrict__ vT,
                 unsigned short* __restrict__ ctx) {
  constexpr int QSTR = 72;   // 64 + 8 pad  (TDM: 32 DWORDs + 4 pad)
  constexpr int VSTR = 136;  // 128 + 8 pad (TDM: 64 DWORDs + 4 pad)
  constexpr int PSTR = 136;
  __shared__ __align__(16) unsigned short Qs[128 * QSTR];
  __shared__ __align__(16) unsigned short Ks[128 * QSTR];
  __shared__ __align__(16) unsigned short VTs[64 * VSTR];  // [d][kk]
  __shared__ __align__(16) unsigned short Ps[128 * PSTR];

  const int tid  = threadIdx.x;
  const int wave = tid >> 5, lane = tid & 31;
  const int g = lane >> 4, c = lane & 15;
  const int q0 = blockIdx.x * 128;
  const int bh = blockIdx.y;
  const size_t head_base  = (size_t)bh * SEQ * DK;
  const size_t vhead_base = (size_t)bh * DK * SEQ;

  // ---- Q tile (once)
#if USE_TDM
  if (wave == 2) {
    tdm_load_2d(lds_addr_of(Qs), qb + head_base + (size_t)q0 * DK,
                /*tile_x=*/DK, /*tile_y=*/128, /*stride=*/DK, /*pi=32dw*/4, /*pa=4dw*/3);
    __builtin_amdgcn_s_wait_tensorcnt(0);
  }
#else
  {
    const int r = tid >> 1, hh = tid & 1;
    const unsigned short* qp = qb + head_base + (size_t)(q0 + r) * DK + 32 * hh;
    unsigned short* qs = Qs + r * QSTR + 32 * hh;
#pragma unroll
    for (int j = 0; j < 4; ++j) *(u32x4*)(qs + 8 * j) = *(const u32x4*)(qp + 8 * j);
  }
#endif

  float run_max[8], run_sum[8];
#pragma unroll
  for (int i = 0; i < 8; ++i) { run_max[i] = -1e30f; run_sum[i] = 0.f; }
  F32Acc oacc[4];
#pragma unroll
  for (int dt = 0; dt < 4; ++dt)
#pragma unroll
    for (int i = 0; i < 8; ++i) oacc[dt].f[i] = 0.f;

  for (int kb0 = 0; kb0 < SEQ; kb0 += 128) {
    // ---- K tile + V^T tile
#if USE_TDM
    if (wave == 0) {
      tdm_load_2d(lds_addr_of(Ks), kbuf + head_base + (size_t)kb0 * DK,
                  DK, 128, DK, 4, 3);
      __builtin_amdgcn_s_wait_tensorcnt(0);
    }
    if (wave == 1) {
      tdm_load_2d(lds_addr_of(VTs), vT + vhead_base + kb0,
                  /*tile_x=*/128, /*tile_y=*/DK, /*stride=*/SEQ, /*pi=64dw*/5, /*pa=4dw*/3);
      __builtin_amdgcn_s_wait_tensorcnt(0);
    }
#else
    {
      const int r = tid >> 1, hh = tid & 1;
      const unsigned short* kp = kbuf + head_base + (size_t)(kb0 + r) * DK + 32 * hh;
      unsigned short* ks = Ks + r * QSTR + 32 * hh;
#pragma unroll
      for (int j = 0; j < 4; ++j) *(u32x4*)(ks + 8 * j) = *(const u32x4*)(kp + 8 * j);
      const int vr = tid >> 2, vq = tid & 3;  // 64 rows x 4 chunks
      const unsigned short* vp = vT + vhead_base + (size_t)vr * SEQ + kb0 + 32 * vq;
      unsigned short* vs = VTs + vr * VSTR + 32 * vq;
#pragma unroll
      for (int j = 0; j < 4; ++j) *(u32x4*)(vs + 8 * j) = *(const u32x4*)(vp + 8 * j);
    }
#endif
    __syncthreads();

    // ---- S = Q @ K^T (wave owns rows 16*wave..+15)
    F32Acc sc[8];
#pragma unroll
    for (int nt = 0; nt < 8; ++nt)
#pragma unroll
      for (int i = 0; i < 8; ++i) sc[nt].f[i] = 0.f;
#pragma unroll
    for (int kt = 0; kt < 2; ++kt) {
      BF16Frag af = load_frag(Qs + 16 * wave * QSTR + 32 * kt, QSTR, lane);
#pragma unroll
      for (int nt = 0; nt < 8; ++nt) {
        BF16Frag bk = load_frag(Ks + 16 * nt * QSTR + 32 * kt, QSTR, lane);
        sc[nt].v = wmma_bf16(af, bk, sc[nt].v);
      }
    }

    // ---- online softmax; row = 16*wave + 8*g + i; scale = 1/8
    float newm[8], fac[8], lsum[8];
#pragma unroll
    for (int i = 0; i < 8; ++i) {
      float m = -1e30f;
#pragma unroll
      for (int nt = 0; nt < 8; ++nt) m = fmaxf(m, sc[nt].f[i]);
      m *= 0.125f;
      m = fmaxf(m, __shfl_xor(m, 1));
      m = fmaxf(m, __shfl_xor(m, 2));
      m = fmaxf(m, __shfl_xor(m, 4));
      m = fmaxf(m, __shfl_xor(m, 8));
      newm[i] = fmaxf(run_max[i], m);
      fac[i]  = __expf(run_max[i] - newm[i]);
      lsum[i] = 0.f;
    }
#pragma unroll
    for (int nt = 0; nt < 8; ++nt)
#pragma unroll
      for (int i = 0; i < 8; ++i) {
        const float p = __expf(sc[nt].f[i] * 0.125f - newm[i]);
        lsum[i] += p;
        Ps[(16 * wave + 8 * g + i) * PSTR + 16 * nt + c] = f2bf(p);
      }
#pragma unroll
    for (int i = 0; i < 8; ++i) {
      float s = lsum[i];
      s += __shfl_xor(s, 1);
      s += __shfl_xor(s, 2);
      s += __shfl_xor(s, 4);
      s += __shfl_xor(s, 8);
      run_sum[i] = run_sum[i] * fac[i] + s;
      run_max[i] = newm[i];
    }
#pragma unroll
    for (int dt = 0; dt < 4; ++dt)
#pragma unroll
      for (int i = 0; i < 8; ++i) oacc[dt].f[i] *= fac[i];
    __syncthreads();

    // ---- O += P @ V
#pragma unroll
    for (int kt = 0; kt < 4; ++kt) {
      BF16Frag ap = load_frag(Ps + 16 * wave * PSTR + 32 * kt, PSTR, lane);
#pragma unroll
      for (int dt = 0; dt < 4; ++dt) {
        BF16Frag bv = load_frag(VTs + 16 * dt * VSTR + 32 * kt, VSTR, lane);
        oacc[dt].v = wmma_bf16(ap, bv, oacc[dt].v);
      }
    }
    __syncthreads();
  }

  // ---- normalize + write context [B,S,DMODEL]
  const int b = bh >> 4, h = bh & 15;
#pragma unroll
  for (int i = 0; i < 8; ++i) {
    const float inv = 1.0f / run_sum[i];
    const int srow = q0 + 16 * wave + 8 * g + i;
    unsigned short* cp = ctx + ((size_t)(b * SEQ + srow)) * DMODEL + h * DK;
#pragma unroll
    for (int dt = 0; dt < 4; ++dt)
      cp[16 * dt + c] = f2bf(oacc[dt].f[i] * inv);
  }
}

// ---------------------------------------------------------------- launch
extern "C" void kernel_launch(void* const* d_in, const int* in_sizes, int n_in,
                              void* d_out, int out_size, void* d_ws, size_t ws_size,
                              hipStream_t stream) {
  const float* Q  = (const float*)d_in[0];
  const float* K  = (const float*)d_in[1];
  const float* V  = (const float*)d_in[2];
  const float* Wq = (const float*)d_in[3];
  const float* bq = (const float*)d_in[4];
  const float* Wk = (const float*)d_in[5];
  const float* bk = (const float*)d_in[6];
  const float* Wv = (const float*)d_in[7];
  const float* bv = (const float*)d_in[8];
  const float* Wo = (const float*)d_in[9];
  const float* bo = (const float*)d_in[10];
  float* out = (float*)d_out;

  unsigned short* ws = (unsigned short*)d_ws;
  const size_t WELT = (size_t)DMODEL * DMODEL;
  const size_t QKV  = (size_t)BATCH * NHEADS * SEQ * DK;
  unsigned short* Wq_bf = ws;
  unsigned short* Wk_bf = Wq_bf + WELT;
  unsigned short* Wv_bf = Wk_bf + WELT;
  unsigned short* Wo_bf = Wv_bf + WELT;
  unsigned short* qb    = Wo_bf + WELT;
  unsigned short* kbf   = qb + QKV;
  unsigned short* vTbf  = kbf + QKV;   // [B,H,DK,S]
  unsigned short* ctx   = vTbf + QKV;

  const dim3 blk(256);
  const dim3 cvt_grid((unsigned)(WELT / 1024));
  cvt_bf16_kernel<<<cvt_grid, blk, 0, stream>>>(Wq, Wq_bf, (int)WELT);
  cvt_bf16_kernel<<<cvt_grid, blk, 0, stream>>>(Wk, Wk_bf, (int)WELT);
  cvt_bf16_kernel<<<cvt_grid, blk, 0, stream>>>(Wv, Wv_bf, (int)WELT);
  cvt_bf16_kernel<<<cvt_grid, blk, 0, stream>>>(Wo, Wo_bf, (int)WELT);

  const dim3 ggrid(DMODEL / 128, (BATCH * SEQ) / 128);  // (8, 32)
  gemm_kernel<true,  1><<<ggrid, blk, 0, stream>>>(Q, Wq_bf, bq, qb);
  gemm_kernel<true,  1><<<ggrid, blk, 0, stream>>>(K, Wk_bf, bk, kbf);
  gemm_kernel<true,  2><<<ggrid, blk, 0, stream>>>(V, Wv_bf, bv, vTbf);

  attn_kernel<<<dim3(SEQ / 128, BATCH * NHEADS), blk, 0, stream>>>(qb, kbf, vTbf, ctx);

  gemm_kernel<false, 0><<<ggrid, blk, 0, stream>>>(ctx, Wo_bf, bo, out);
}
